// DarkCapsuleNet_72464688218166
// MI455X (gfx1250) — matrix operand order repaired and checked by reading, not compiled, not run on hardware
//
#include <hip/hip_runtime.h>
#include <hip/hip_bf16.h>
#include <math.h>
#include <stdint.h>

typedef __bf16 bf16_t;
typedef __attribute__((ext_vector_type(16))) __bf16 v16bf;
typedef __attribute__((ext_vector_type(8)))  __bf16 v8bf;
typedef __attribute__((ext_vector_type(8)))  float  v8f;
typedef __attribute__((ext_vector_type(4)))  unsigned int v4u;
typedef __attribute__((ext_vector_type(8)))  int v8i;
typedef __attribute__((ext_vector_type(4)))  int v4i;

union Frag { v16bf v; v8bf h[2]; };

// ---------------- TDM helper: 2D tile (rows x cols bf16, row stride in elems) -> LDS
#if __has_builtin(__builtin_amdgcn_tensor_load_to_lds)
#define HAVE_TDM 1
__device__ __forceinline__ void tdm_load_2d(unsigned lds_off, const bf16_t* gsrc,
                                            int cols, int rows, int rowstride_elems) {
  unsigned long long ga = (unsigned long long)(uintptr_t)gsrc;
  v4u g0; v8i g1; v4i gz;
  // D# group0: [1:0]=count=1 (user, valid), [63:32]=lds_addr, [120:64]=global_addr, [127:126]=type=2
  g0[0] = 1u;
  g0[1] = lds_off;
  g0[2] = (unsigned)ga;
  g0[3] = (unsigned)((ga >> 32) & 0x01ffffffu) | (2u << 30);
  // D# group1: data_size=1 (2B); tensor_dim0=cols; tensor_dim1=rows;
  //            tile_dim0=cols; tile_dim1=rows; tile_dim2=0; dim0_stride=rowstride
  unsigned td0 = (unsigned)cols;
  unsigned td1 = (unsigned)rows;
  unsigned long long st0 = (unsigned long long)rowstride_elems;
  g1[0] = (int)(1u << 16);                                   // data_size=2B, mask=0
  g1[1] = (int)((td0 & 0xffffu) << 16);                      // tensor_dim0[15:0] @63:48
  g1[2] = (int)(((td0 >> 16) & 0xffffu) | ((td1 & 0xffffu) << 16));
  g1[3] = (int)(((td1 >> 16) & 0xffffu) | ((td0 & 0xffffu) << 16)); // tile_dim0 @127:112
  g1[4] = (int)(td1 & 0xffffu);                              // tile_dim1 @143:128, tile_dim2=0
  g1[5] = (int)(unsigned)(st0 & 0xffffffffu);                // dim0_stride lo
  g1[6] = (int)((st0 >> 32) & 0xffffu);                      // dim0_stride hi; dim1_stride=0
  g1[7] = 0;
  gz[0] = 0; gz[1] = 0; gz[2] = 0; gz[3] = 0;
#if __clang_major__ >= 23
  v8i g4; g4[0]=0; g4[1]=0; g4[2]=0; g4[3]=0; g4[4]=0; g4[5]=0; g4[6]=0; g4[7]=0;
  __builtin_amdgcn_tensor_load_to_lds(g0, g1, gz, gz, g4, 0);
#else
  __builtin_amdgcn_tensor_load_to_lds(g0, g1, gz, gz, 0);
#endif
  __builtin_amdgcn_s_wait_tensorcnt((short)0);
}
#else
#define HAVE_TDM 0
#endif

// ---------------- elementwise / prep kernels ----------------

__global__ void k_zero_f32(float* __restrict__ p, int n) {
  int i = blockIdx.x * blockDim.x + threadIdx.x;
  if (i < n) p[i] = 0.f;
}

// OIHW f32 -> (O, R, S, I) bf16  (K index = (r*R+s)*I + i, contiguous per row)
__global__ void k_prep_w(const float* __restrict__ w, bf16_t* __restrict__ wt,
                         int O, int I, int R) {
  int idx = blockIdx.x * blockDim.x + threadIdx.x;
  int total = O * I * R * R;
  if (idx >= total) return;
  int i = idx % I; int t = idx / I;
  int s = t % R; t /= R;
  int r = t % R; int o = t / R;
  wt[idx] = (bf16_t)w[((o * I + i) * R + r) * R + s];
}

// ---------------- conv1: direct 3x3, Cin=3, NCHW f32 -> NHWC bf16 raw
__global__ void k_conv1(const float* __restrict__ x, const float* __restrict__ w,
                        const float* __restrict__ bias, bf16_t* __restrict__ out) {
  int gid = blockIdx.x * blockDim.x + threadIdx.x;   // 8*128*128*128
  if (gid >= 8 * 128 * 128 * 128) return;
  int co = gid & 127; int t = gid >> 7;
  int wo = t & 127; t >>= 7;
  int ho = t & 127; int b = t >> 7;
  float acc = bias[co];
  for (int ci = 0; ci < 3; ++ci)
    for (int r = 0; r < 3; ++r) {
      int hi = ho + r - 1;
      if (hi < 0 || hi >= 128) continue;
      for (int s = 0; s < 3; ++s) {
        int wi = wo + s - 1;
        if (wi < 0 || wi >= 128) continue;
        acc += x[((b * 3 + ci) * 128 + hi) * 128 + wi] * w[((co * 3 + ci) * 3 + r) * 3 + s];
      }
    }
  out[gid] = (bf16_t)acc;
}

// ---------------- BN stats: one block per channel, deterministic tree reduce
__global__ __launch_bounds__(256) void k_bn_stats(const bf16_t* __restrict__ x,
                                                  float* __restrict__ stats,
                                                  int spatial, int C) {
  int c = blockIdx.x;
  int tid = threadIdx.x;
  float s0 = 0.f, s1 = 0.f;
  for (int sp = tid; sp < spatial; sp += 256) {
    float v = (float)x[(size_t)sp * C + c];
    s0 += v; s1 += v * v;
  }
  __shared__ float r0[256];
  __shared__ float r1[256];
  r0[tid] = s0; r1[tid] = s1;
  __syncthreads();
  for (int w = 128; w > 0; w >>= 1) {
    if (tid < w) { r0[tid] += r0[tid + w]; r1[tid] += r1[tid + w]; }
    __syncthreads();
  }
  if (tid == 0) { stats[c] = r0[0]; stats[C + c] = r1[0]; }
}

// ---------------- BN normalize + LeakyReLU, in place (bf16)
__global__ void k_bn_lrelu(bf16_t* __restrict__ x, const float* __restrict__ stats,
                           const float* __restrict__ gamma, const float* __restrict__ beta,
                           int total, int C) {
  int i = blockIdx.x * blockDim.x + threadIdx.x;
  if (i >= total) return;
  int c = i & (C - 1);                 // C is a power of two
  float cnt = (float)(total / C);
  float m = stats[c] / cnt;
  float var = stats[C + c] / cnt - m * m;
  float v = ((float)x[i] - m) * rsqrtf(var + 1e-5f);
  v = v * gamma[c] + beta[c];
  v = (v >= 0.f) ? v : 0.1f * v;
  x[i] = (bf16_t)v;
}

// ---------------- implicit-GEMM conv via WMMA bf16 (NHWC in/out, weights (O,R,S,I))
// wave: 32(M=Cout) x 32(N=spatial) tile; block: 8 waves -> 64M x 128N.
// Per (r,s) tap: TDM DMAs the 64xCin weight tile into LDS; A-frags come from LDS.
__global__ __launch_bounds__(256) void k_conv_wmma(
    const bf16_t* __restrict__ in, const bf16_t* __restrict__ wt,
    const float* __restrict__ bias, bf16_t* __restrict__ out,
    int H, int W, int Cin, int Cout,
    int Ho, int Wo, int R, int stride, int pad)
{
  __shared__ __align__(16) bf16_t sA[64 * 256];   // 32 KB max (Cin <= 256)
  const int lane  = threadIdx.x & 31;
  const int wave  = threadIdx.x >> 5;
  const int l16   = lane & 15;
  const int lhalf = lane >> 4;
  const int K     = R * R * Cin;

  const int mblk = blockIdx.y * 64;
  const int mw   = (wave >> 2) * 32;              // M offset within block tile
  const int m0   = mblk + mw;
  const int n0   = blockIdx.x * 128 + (wave & 3) * 32;

  v8f acc[2][2];
#pragma unroll
  for (int a = 0; a < 2; ++a)
#pragma unroll
    for (int b = 0; b < 2; ++b)
#pragma unroll
      for (int q = 0; q < 8; ++q) acc[a][b][q] = 0.f;

  // Decode N (spatial) per dn: lane holds N=l16; K chunks at +lhalf*8 and +lhalf*8+16
  int nidx[2], bb[2], ho[2], wo[2];
#pragma unroll
  for (int dn = 0; dn < 2; ++dn) {
    int n = n0 + dn * 16 + l16;
    nidx[dn] = n;
    wo[dn] = n % Wo; int t = n / Wo;
    ho[dn] = t % Ho; bb[dn] = t / Ho;
  }

  Frag zf;
#pragma unroll
  for (int q = 0; q < 16; ++q) zf.v[q] = (bf16_t)0.0f;

#if HAVE_TDM
  const unsigned lds_base = (unsigned)(uintptr_t)(&sA[0]);  // low 32 bits = LDS offset
#endif

  for (int r = 0; r < R; ++r) {
    for (int s = 0; s < R; ++s) {
      const bf16_t* bptr[2]; bool ok[2];
#pragma unroll
      for (int dn = 0; dn < 2; ++dn) {
        int hi = ho[dn] * stride + r - pad;
        int wi = wo[dn] * stride + s - pad;
        ok[dn] = (hi >= 0) & (hi < H) & (wi >= 0) & (wi < W);
        int hic = hi < 0 ? 0 : (hi >= H ? H - 1 : hi);
        int wic = wi < 0 ? 0 : (wi >= W ? W - 1 : wi);
        bptr[dn] = in + (((size_t)bb[dn] * H + hic) * W + wic) * Cin + lhalf * 8;
      }
      if (ok[0]) __builtin_prefetch(bptr[0], 0, 1);   // global_prefetch_b8
      const int kbase = (r * R + s) * Cin;

      __syncthreads();   // previous tile fully consumed
#if HAVE_TDM
      if (threadIdx.x < 32)   // wave 0 issues the DMA (TDM ignores EXEC) + drains it
        tdm_load_2d(lds_base, wt + (size_t)mblk * K + kbase, Cin, 64, K);
#else
      for (int idx = threadIdx.x; idx < (64 * Cin) >> 3; idx += 256) {
        int row = idx / (Cin >> 3);
        int c8  = idx - row * (Cin >> 3);
        *((v8bf*)sA + idx) =
            *(const v8bf*)(wt + (size_t)(mblk + row) * K + kbase + c8 * 8);
      }
#endif
      __syncthreads();   // tile visible to all waves

      for (int c0 = 0; c0 < Cin; c0 += 32) {
        Frag a[2], bfr[2];
#pragma unroll
        for (int dm = 0; dm < 2; ++dm) {
          // A layout: lane holds M=l16; K chunks at +lhalf*8 and +lhalf*8+16
          const bf16_t* p = sA + (size_t)(mw + dm * 16 + l16) * Cin + c0 + lhalf * 8;
          a[dm].h[0] = *(const v8bf*)(p);         // ds_load_b128
          a[dm].h[1] = *(const v8bf*)(p + 16);
        }
#pragma unroll
        for (int dn = 0; dn < 2; ++dn) {
          if (ok[dn]) {
            const bf16_t* p = bptr[dn] + c0;
            bfr[dn].h[0] = *(const v8bf*)(p);
            bfr[dn].h[1] = *(const v8bf*)(p + 16);
          } else {
            bfr[dn] = zf;
          }
        }
#pragma unroll
        for (int dm = 0; dm < 2; ++dm)
#pragma unroll
          for (int dn = 0; dn < 2; ++dn)
            acc[dm][dn] = __builtin_amdgcn_wmma_f32_16x16x32_bf16(
                false, a[dm].v, false, bfr[dn].v, (short)0, acc[dm][dn], false, false);
      }
    }
  }

  // epilogue: C/D layout — lane holds N=l16; VGPR v -> M = v + 8*lhalf
#pragma unroll
  for (int dm = 0; dm < 2; ++dm) {
#pragma unroll
    for (int dn = 0; dn < 2; ++dn) {
      int n = nidx[dn];
      int cbase = m0 + dm * 16 + lhalf * 8;
      bf16_t* op = out + (size_t)n * Cout + cbase;
#pragma unroll
      for (int v = 0; v < 8; ++v) {
        float val = acc[dm][dn][v] + bias[cbase + v];
        op[v] = (bf16_t)val;
      }
    }
  }
}

// ---------------- capsule gather: A5 NHWC bf16 (8,16,16,256) -> caps f32 (128,512,8)
__global__ void k_caps(const bf16_t* __restrict__ h, float* __restrict__ caps) {
  int idx = blockIdx.x * blockDim.x + threadIdx.x;   // 128*512*8
  if (idx >= 524288) return;
  int i = idx & 7; int t = idx >> 3;
  int n = t & 511; int bbv = t >> 9;
  int g = bbv >> 3, b = bbv & 7;
  int i2 = n >> 7; int i4 = (n >> 5) & 3; int c8 = n & 31;
  int c = c8 * 8 + i;
  int hh = i2 * 4 + (g >> 2);
  int ww = (g & 3) * 4 + i4;
  caps[idx] = (float)h[(((size_t)b * 16 + hh) * 16 + ww) * 256 + c];
}

// ---------------- routing: softmax over k (43) per (bb,n)
__global__ void k_softmax(const float* __restrict__ logits, float* __restrict__ probs,
                          int rows) {
  int row = blockIdx.x * blockDim.x + threadIdx.x;   // 128*512
  if (row >= rows) return;
  const float* L = logits + (size_t)row * 43;
  float* P = probs + (size_t)row * 43;
  float mx = L[0];
  for (int k = 1; k < 43; ++k) mx = fmaxf(mx, L[k]);
  float e[43]; float sum = 0.f;
  for (int k = 0; k < 43; ++k) { e[k] = __expf(L[k] - mx); sum += e[k]; }
  float inv = 1.f / sum;
  for (int k = 0; k < 43; ++k) P[k] = e[k] * inv;
}

// s[bb,k,o] = sum_n probs[bb,n,k] * (sum_i caps[bb,n,i]*rw[n,k,i,o]); then squash -> outs
__global__ __launch_bounds__(256) void k_route_s(
    const float* __restrict__ caps, const float* __restrict__ rw,
    const float* __restrict__ probs, float* __restrict__ outs) {
  int bbv = blockIdx.x;   // 128
  __shared__ float scaps[512 * 8];
  __shared__ float s[43 * 21];
  for (int i = threadIdx.x; i < 512 * 8; i += 256)
    scaps[i] = caps[(size_t)bbv * 4096 + i];
  __syncthreads();
  for (int ko = threadIdx.x; ko < 903; ko += 256) {
    int k = ko / 21, o = ko % 21;
    float acc = 0.f;
    for (int n = 0; n < 512; ++n) {
      const float* cv = scaps + n * 8;
      const float* rp = rw + (((size_t)n * 43 + k) * 8) * 21 + o;
      float p = 0.f;
#pragma unroll
      for (int i = 0; i < 8; ++i) p += cv[i] * rp[i * 21];
      acc += probs[((size_t)bbv * 512 + n) * 43 + k] * p;
    }
    s[ko] = acc;
  }
  __syncthreads();
  for (int k = threadIdx.x; k < 43; k += 256) {
    float sn = 0.f;
    for (int o = 0; o < 21; ++o) { float v = s[k * 21 + o]; sn += v * v; }
    float sc = sn / (1.f + sn) * rsqrtf(sn);
    for (int o = 0; o < 21; ++o)
      outs[((size_t)bbv * 43 + k) * 21 + o] = s[k * 21 + o] * sc;
  }
}

// logits[bb,n,k] += sum_o prior(bb,n,k,o) * outs[bb,k,o]
__global__ void k_route_update(const float* __restrict__ caps, const float* __restrict__ rw,
                               const float* __restrict__ outs, float* __restrict__ logits,
                               int total) {
  int idx = blockIdx.x * blockDim.x + threadIdx.x;   // 128*512*43
  if (idx >= total) return;
  int k = idx % 43; int t = idx / 43;
  int n = t % 512; int bbv = t / 512;
  const float* cv = caps + ((size_t)bbv * 512 + n) * 8;
  const float* rp = rw + (((size_t)n * 43 + k) * 8) * 21;
  const float* ov = outs + ((size_t)bbv * 43 + k) * 21;
  float acc = 0.f;
  for (int o = 0; o < 21; ++o) {
    float p = 0.f;
#pragma unroll
    for (int i = 0; i < 8; ++i) p += cv[i] * rp[i * 21 + o];
    acc += p * ov[o];
  }
  logits[idx] += acc;
}

// out[b,g1,g2,k,o] = outs[(g1*4+g2)*8+b, k, o]
__global__ void k_final(const float* __restrict__ outs, float* __restrict__ out) {
  int idx = blockIdx.x * blockDim.x + threadIdx.x;   // 8*4*4*43*21
  if (idx >= 115584) return;
  int o = idx % 21; int t = idx / 21;
  int k = t % 43; t /= 43;
  int g2 = t % 4; t /= 4;
  int g1 = t % 4; int b = t / 4;
  int bbv = (g1 * 4 + g2) * 8 + b;
  out[idx] = outs[((size_t)bbv * 43 + k) * 21 + o];
}

// ---------------- host orchestration ----------------
extern "C" void kernel_launch(void* const* d_in, const int* in_sizes, int n_in,
                              void* d_out, int out_size, void* d_ws, size_t ws_size,
                              hipStream_t stream) {
  (void)in_sizes; (void)n_in; (void)out_size; (void)ws_size;
  const float* x   = (const float*)d_in[0];
  const float* c1w = (const float*)d_in[1];
  const float* c1b = (const float*)d_in[2];
  const float* g1  = (const float*)d_in[3];
  const float* b1  = (const float*)d_in[4];
  const float* c2w = (const float*)d_in[5];
  const float* c2b = (const float*)d_in[6];
  const float* g2  = (const float*)d_in[7];
  const float* b2  = (const float*)d_in[8];
  const float* c3w = (const float*)d_in[9];
  const float* c3b = (const float*)d_in[10];
  const float* g3  = (const float*)d_in[11];
  const float* b3  = (const float*)d_in[12];
  const float* c4w = (const float*)d_in[13];
  const float* c4b = (const float*)d_in[14];
  const float* g4  = (const float*)d_in[15];
  const float* b4  = (const float*)d_in[16];
  const float* c5w = (const float*)d_in[17];
  const float* c5b = (const float*)d_in[18];
  const float* g5  = (const float*)d_in[19];
  const float* b5  = (const float*)d_in[20];
  const float* rw  = (const float*)d_in[21];
  float* out = (float*)d_out;

  char* ws = (char*)d_ws;
  size_t off = 0;
  auto alloc = [&](size_t bytes) -> void* {
    void* p = (void*)(ws + off);
    off += (bytes + 255) & ~(size_t)255;
    return p;
  };
  float*  stats  = (float*) alloc(512 * 4);
  bf16_t* A1     = (bf16_t*)alloc((size_t)16777216 * 2);   // (8,128,128,128) NHWC
  bf16_t* A2     = (bf16_t*)alloc((size_t)33554432 * 2);   // (8,128,128,256)
  bf16_t* A3     = (bf16_t*)alloc((size_t)2097152 * 2);    // (8,64,64,64)
  bf16_t* A4     = (bf16_t*)alloc((size_t)1048576 * 2);    // (8,32,32,128)
  bf16_t* A5     = (bf16_t*)alloc((size_t)524288 * 2);     // (8,16,16,256)
  bf16_t* W2t    = (bf16_t*)alloc((size_t)294912 * 2);     // (256,3,3,128)
  bf16_t* W3t    = (bf16_t*)alloc((size_t)262144 * 2);     // (64,4,4,256)
  bf16_t* W4t    = (bf16_t*)alloc((size_t)131072 * 2);     // (128,4,4,64)
  bf16_t* W5t    = (bf16_t*)alloc((size_t)524288 * 2);     // (256,4,4,128)
  float*  caps   = (float*) alloc((size_t)524288 * 4);     // (128,512,8)
  float*  logits = (float*) alloc((size_t)2818048 * 4);    // (128,512,43)
  float*  probs  = (float*) alloc((size_t)2818048 * 4);
  float*  outsb  = (float*) alloc((size_t)115584 * 4);     // (128,43,21)

  const int TB = 256;

  // weight prep (OIHW f32 -> ORSI bf16)
  k_prep_w<<<(256 * 128 * 9 + TB - 1) / TB, TB, 0, stream>>>(c2w, W2t, 256, 128, 3);
  k_prep_w<<<(64 * 256 * 16 + TB - 1) / TB, TB, 0, stream>>>(c3w, W3t, 64, 256, 4);
  k_prep_w<<<(128 * 64 * 16 + TB - 1) / TB, TB, 0, stream>>>(c4w, W4t, 128, 64, 4);
  k_prep_w<<<(256 * 128 * 16 + TB - 1) / TB, TB, 0, stream>>>(c5w, W5t, 256, 128, 4);

  // layer 1 (direct conv) + BN + lrelu
  k_conv1<<<16777216 / TB, TB, 0, stream>>>(x, c1w, c1b, A1);
  k_bn_stats<<<128, 256, 0, stream>>>(A1, stats, 131072, 128);
  k_bn_lrelu<<<16777216 / TB, TB, 0, stream>>>(A1, stats, g1, b1, 16777216, 128);

  // layer 2: WMMA GEMM  M=256 N=131072 K=1152
  k_conv_wmma<<<dim3(1024, 4), 256, 0, stream>>>(A1, W2t, c2b, A2,
      128, 128, 128, 256, 128, 128, 3, 1, 1);
  k_bn_stats<<<256, 256, 0, stream>>>(A2, stats, 131072, 256);
  k_bn_lrelu<<<33554432 / TB, TB, 0, stream>>>(A2, stats, g2, b2, 33554432, 256);

  // layer 3: M=64 N=32768 K=4096
  k_conv_wmma<<<dim3(256, 1), 256, 0, stream>>>(A2, W3t, c3b, A3,
      128, 128, 256, 64, 64, 64, 4, 2, 1);
  k_bn_stats<<<64, 256, 0, stream>>>(A3, stats, 32768, 64);
  k_bn_lrelu<<<2097152 / TB, TB, 0, stream>>>(A3, stats, g3, b3, 2097152, 64);

  // layer 4: M=128 N=8192 K=1024
  k_conv_wmma<<<dim3(64, 2), 256, 0, stream>>>(A3, W4t, c4b, A4,
      64, 64, 64, 128, 32, 32, 4, 2, 1);
  k_bn_stats<<<128, 256, 0, stream>>>(A4, stats, 8192, 128);
  k_bn_lrelu<<<1048576 / TB, TB, 0, stream>>>(A4, stats, g4, b4, 1048576, 128);

  // layer 5: M=256 N=2048 K=2048
  k_conv_wmma<<<dim3(16, 4), 256, 0, stream>>>(A4, W5t, c5b, A5,
      32, 32, 128, 256, 16, 16, 4, 2, 1);
  k_bn_stats<<<256, 256, 0, stream>>>(A5, stats, 2048, 256);
  k_bn_lrelu<<<524288 / TB, TB, 0, stream>>>(A5, stats, g5, b5, 524288, 256);

  // capsules + dynamic routing (priors recomputed on the fly)
  k_caps<<<524288 / TB, TB, 0, stream>>>(A5, caps);
  k_zero_f32<<<(2818048 + TB - 1) / TB, TB, 0, stream>>>(logits, 2818048);

  for (int it = 0; it < 3; ++it) {
    k_softmax<<<(65536 + TB - 1) / TB, TB, 0, stream>>>(logits, probs, 65536);
    k_route_s<<<128, 256, 0, stream>>>(caps, rw, probs, outsb);
    if (it != 2)
      k_route_update<<<(2818048 + TB - 1) / TB, TB, 0, stream>>>(caps, rw, outsb, logits, 2818048);
  }
  k_final<<<(115584 + TB - 1) / TB, TB, 0, stream>>>(outsb, out);
}
